// Generator_38800734552637
// MI455X (gfx1250) — compile-verified, matrix operand order checked by reading, not run on hardware
//
#include <hip/hip_runtime.h>
#include <math.h>

typedef _Float16 half8   __attribute__((ext_vector_type(8)));
typedef _Float16 half16  __attribute__((ext_vector_type(16)));
typedef float    floatx8 __attribute__((ext_vector_type(8)));

#define LSTR 40                      // LDS row stride in halves: 32 data + 8 pad (80B, keeps 16B align)
#define SQRT2F 1.41421356237309515f

// ---- constant filter taps -------------------------------------------------
__device__ __constant__ float K_BLUR[16] = {   // outer([1,3,3,1])/64 * 4
  0.0625f, 0.1875f, 0.1875f, 0.0625f,
  0.1875f, 0.5625f, 0.5625f, 0.1875f,
  0.1875f, 0.5625f, 0.5625f, 0.1875f,
  0.0625f, 0.1875f, 0.1875f, 0.0625f };
__device__ __constant__ float K_HAAR[4][4] = { // ll, lh, hl, hh (2x2 row-major)
  { 0.5f,  0.5f,  0.5f,  0.5f},
  {-0.5f, -0.5f,  0.5f,  0.5f},
  {-0.5f,  0.5f, -0.5f,  0.5f},
  { 0.5f, -0.5f, -0.5f,  0.5f} };

__device__ inline half16 frag_from(const _Float16* base) {
  half8 lo = *(const half8*)base;
  half8 hi = *(const half8*)(base + 16);
  return __builtin_shufflevector(lo, hi, 0,1,2,3,4,5,6,7,8,9,10,11,12,13,14,15);
}

// ---- mapping network ------------------------------------------------------
__global__ void pixel_norm_k(const float* __restrict__ x, float* __restrict__ o) {
  __shared__ float red[256];
  const int b = blockIdx.x, t = threadIdx.x;
  float a0 = x[b*512 + t], a1 = x[b*512 + 256 + t];
  red[t] = a0*a0 + a1*a1;
  __syncthreads();
  for (int s = 128; s > 0; s >>= 1) { if (t < s) red[t] += red[t+s]; __syncthreads(); }
  float r = rsqrtf(red[0] * (1.0f/512.0f) + 1e-8f);
  o[b*512 + t] = a0*r; o[b*512 + 256 + t] = a1*r;
}

__global__ void prep_mapw_k(const float* __restrict__ w, _Float16* __restrict__ o) {
  long t = (long)blockIdx.x*256 + threadIdx.x;
  if (t >= 262144) return;
  o[t] = (_Float16)(w[t] * 4.4194173824159216e-4f);   // 0.01/sqrt(512)
}

// out^T GEMM: M=feature(512), N=batch(16, 8 valid), K=512, f16 WMMA
__global__ void __launch_bounds__(256)
map_layer_k(const float* __restrict__ xin, const _Float16* __restrict__ wl,
            const float* __restrict__ bias, float* __restrict__ xout) {
  __shared__ __align__(16) _Float16 Xt[16*LSTR];
  const int tid = threadIdx.x, wave = tid>>5, lane = tid&31;
  const int g = lane>>4, mr = lane&15;
  const int rowB = blockIdx.x*128 + wave*16;
  floatx8 acc = {0.f,0.f,0.f,0.f,0.f,0.f,0.f,0.f};
  for (int c0 = 0; c0 < 512; c0 += 32) {
    #pragma unroll
    for (int e = 0; e < 2; ++e) {
      int idx = tid + e*256; int bb = idx>>5, kk = idx&31;
      _Float16 v = (_Float16)0.f;
      if (bb < 8) v = (_Float16)xin[bb*512 + c0 + kk];
      Xt[bb*LSTR + kk] = v;
    }
    __syncthreads();
    half16 af = frag_from(wl + (size_t)(rowB + mr)*512 + c0 + g*8);
    half16 bf = frag_from(&Xt[mr*LSTR + g*8]);
    acc = __builtin_amdgcn_wmma_f32_16x16x32_f16(false, af, false, bf, (short)0, acc, false, false);
    __syncthreads();
  }
  if (mr < 8) {
    #pragma unroll
    for (int r = 0; r < 8; ++r) {
      int n = rowB + r + 8*g;
      float v = acc[r] + bias[n]*0.01f;
      v = (v > 0.f ? v : 0.2f*v) * SQRT2F;
      xout[mr*512 + n] = v;
    }
  }
}

// ---- per-conv preparation -------------------------------------------------
// s[b][ci] = wlat[b] . mod_w[ci] / sqrt(512) + mod_b[ci]
__global__ void style_mod_k(const float* __restrict__ wlat, const float* __restrict__ mw,
                            const float* __restrict__ mb, float* __restrict__ s, int Cin) {
  long t = (long)blockIdx.x*256 + threadIdx.x;
  if (t >= (long)8*Cin) return;
  int ci = (int)(t % Cin), b = (int)(t / Cin);
  const float* xr = wlat + (long)b*512;
  const float* wr = mw   + (long)ci*512;
  float acc = 0.f;
  for (int k = 0; k < 512; ++k) acc += xr[k]*wr[k];
  s[t] = acc * 0.044194173824159216f + mb[ci];
}

// wh[tap][co][ci] = f16(scale*w) (tap order flipped when flip=1); w2[co][ci] = sum_tap (scale*w)^2
__global__ void prep_convw_k(const float* __restrict__ w, _Float16* __restrict__ wh,
                             float* __restrict__ w2, int Cin, int Cout, int k, int flip) {
  long t = (long)blockIdx.x*256 + threadIdx.x;
  if (t >= (long)Cout*Cin) return;
  int co = (int)(t / Cin), ci = (int)(t % Cin);
  float scale = rsqrtf((float)(Cin*k*k));
  float s2 = 0.f;
  for (int ky = 0; ky < k; ++ky)
    for (int kx = 0; kx < k; ++kx) {
      float v = scale * w[(((size_t)co*Cin + ci)*k + ky)*k + kx];
      s2 += v*v;
      int ty = flip ? (k-1-ky) : ky, tx = flip ? (k-1-kx) : kx;
      wh[(((size_t)(ty*k+tx))*Cout + co)*Cin + ci] = (_Float16)v;
    }
  w2[t] = s2;
}

__global__ void demod_k(const float* __restrict__ w2, const float* __restrict__ s,
                        float* __restrict__ d, int Cin, int Cout) {
  long t = (long)blockIdx.x*256 + threadIdx.x;
  if (t >= (long)8*Cout) return;
  int co = (int)(t % Cout), b = (int)(t / Cout);
  const float* sr = s  + (long)b*Cin;
  const float* wr = w2 + (long)co*Cin;
  float acc = 0.f;
  for (int ci = 0; ci < Cin; ++ci) { float sv = sr[ci]; acc += wr[ci]*sv*sv; }
  d[t] = rsqrtf(acc + 1e-8f);
}

// xs_h = f16(x * s) ; xbs==0 broadcasts batch (const input)
__global__ void mod_half_k(const float* __restrict__ x, long xbs, const float* __restrict__ s,
                           _Float16* __restrict__ o, int Cin, int HW, long total) {
  long t = (long)blockIdx.x*256 + threadIdx.x;
  if (t >= total) return;
  int p = (int)(t % HW); long r = t / HW; int c = (int)(r % Cin); int b = (int)(r / Cin);
  o[t] = (_Float16)(x[(long)b*xbs + (long)c*HW + p] * s[(long)b*Cin + c]);
}

// zero-dilate 2x (size 2H-1) fused with modulation -> f16
__global__ void dilate_mod_half_k(const float* __restrict__ x, const float* __restrict__ s,
                                  _Float16* __restrict__ o, int Cin, int H, int W, long total) {
  long t = (long)blockIdx.x*256 + threadIdx.x;
  if (t >= total) return;
  int D = 2*H - 1, DW = 2*W - 1;
  int ox = (int)(t % DW); long r = t / DW; int oy = (int)(r % D); r /= D;
  int c = (int)(r % Cin); int b = (int)(r / Cin);
  float v = 0.f;
  if (!(oy & 1) && !(ox & 1))
    v = x[(((long)b*Cin + c)*H + (oy>>1))*W + (ox>>1)] * s[(long)b*Cin + c];
  o[t] = (_Float16)v;
}

// ---- WMMA implicit-GEMM convolution --------------------------------------
// out[b][co][oy][ox] = sum_{tap,ci} wh[tap][co][ci] * xs[b][ci][oy+ky-pad][ox+kx-pad]  (correlation)
// block = 256 thr = 8 waves (2 M x 4 N); per wave a 2x2 grid of 16x16 WMMA C tiles:
//   block tile = 64 cout x (8 rows x 16 cols) pixels; 4 WMMAs per LDS stage, fragment reuse.
// A tile (64x32 f16 weights) is staged with async DMA: one global_load_async_to_lds_b128
// per thread per stage (contiguous on both sides), completed with s_wait_asynccnt before
// the block barrier (barriers wait DScnt, not ASYNCcnt).
__global__ void __launch_bounds__(256)
conv_wmma_k(const _Float16* __restrict__ xs, const _Float16* __restrict__ wtab,
            const float* __restrict__ demod, float* __restrict__ out,
            int Cin, int Cout, int IH, int IW, int OH, int OW, int ks, int pad,
            int tilesX) {
  __shared__ __align__(16) _Float16 At[64*LSTR];    // 64 cout x 32 cin
  __shared__ __align__(16) _Float16 Bt[128*LSTR];   // 128 pixels x 32 cin
  const int tid = threadIdx.x, wave = tid>>5, lane = tid&31;
  const int waveM = wave>>2;          // 0..1
  const int waveN = wave&3;           // 0..3
  const int g = lane>>4, mr = lane&15;
  const int b = blockIdx.z;
  const int coB = blockIdx.y*64;
  const int tileY = (int)blockIdx.x / tilesX;
  const int tileX = (int)blockIdx.x - tileY*tilesX;
  const int oyB = tileY*8, oxB = tileX*16;
  const long np = (long)OH*OW;

  // async A-staging addressing: thread -> (row, 8-half segment); clamp row for Cout<64
  const int aco = tid >> 2, aseg = tid & 3;
  int agco = coB + aco; if (agco >= Cout) agco = Cout - 1;
  const unsigned aLds = (unsigned)(size_t)(&At[aco*LSTR + aseg*8]);

  floatx8 acc[2][2];
  const floatx8 z = {0.f,0.f,0.f,0.f,0.f,0.f,0.f,0.f};
  acc[0][0]=z; acc[0][1]=z; acc[1][0]=z; acc[1][1]=z;

  const int ntap = ks*ks;
  for (int tap = 0; tap < ntap; ++tap) {
    const int ky = tap/ks - pad, kx = tap%ks - pad;
    const _Float16* wt = wtab + (size_t)tap*Cout*Cin;
    for (int c0 = 0; c0 < Cin; c0 += 32) {
      // A tile: one 16B async DMA to LDS per thread (gfx1250 ASYNCcnt path)
      {
        const _Float16* gp = wt + (size_t)agco*Cin + c0 + aseg*8;
        asm volatile("global_load_async_to_lds_b128 %0, %1, off"
                     :: "v"(aLds), "v"(gp) : "memory");
      }
      if (c0 + 32 < Cin) __builtin_prefetch(wt + (size_t)coB*Cin + c0 + 32, 0, 1);
      #pragma unroll
      for (int e = 0; e < 16; ++e) {                // B tile: 128 px x 32 (lanes sweep px: coalesced)
        int idx = tid + e*256; int p = idx & 127, ci = idx >> 7;
        int py = p >> 4, px = p & 15;
        int iy = oyB + py + ky, ix = oxB + px + kx;
        _Float16 v = (_Float16)0.f;
        if ((unsigned)iy < (unsigned)IH && (unsigned)ix < (unsigned)IW)
          v = xs[(((size_t)b*Cin + c0 + ci)*IH + iy)*IW + ix];
        Bt[p*LSTR + ci] = v;
      }
      asm volatile("s_wait_asynccnt 0x0" ::: "memory");   // A DMA landed in LDS
      __syncthreads();
      half16 af[2], bf[2];
      #pragma unroll
      for (int mm = 0; mm < 2; ++mm)
        af[mm] = frag_from(&At[(waveM*32 + mm*16 + mr)*LSTR + g*8]);
      #pragma unroll
      for (int nn = 0; nn < 2; ++nn)
        bf[nn] = frag_from(&Bt[((waveN + nn*4)*16 + mr)*LSTR + g*8]);
      #pragma unroll
      for (int mm = 0; mm < 2; ++mm)
        #pragma unroll
        for (int nn = 0; nn < 2; ++nn)
          acc[mm][nn] = __builtin_amdgcn_wmma_f32_16x16x32_f16(
              false, af[mm], false, bf[nn], (short)0, acc[mm][nn], false, false);
      __syncthreads();
    }
  }

  #pragma unroll
  for (int nn = 0; nn < 2; ++nn) {
    int oy = oyB + waveN + nn*4;
    int ox = oxB + mr;
    if (oy < OH && ox < OW) {
      long n = (long)oy*OW + ox;
      #pragma unroll
      for (int mm = 0; mm < 2; ++mm) {
        #pragma unroll
        for (int r = 0; r < 8; ++r) {
          int m = coB + waveM*32 + mm*16 + r + 8*g;
          if (m < Cout) {
            float v = acc[mm][nn][r];
            if (demod) v *= demod[b*Cout + m];
            out[((size_t)b*Cout + m)*np + n] = v;   // 16-wide coalesced along ox
          }
        }
      }
    }
  }
}

// ---- epilogues & resampling ----------------------------------------------
__global__ void noise_act_k(float* __restrict__ x, const float* __restrict__ noise,
                            const float* __restrict__ nw, const float* __restrict__ cb,
                            int C, int HW, int doAct, long total) {
  long t = (long)blockIdx.x*256 + threadIdx.x;
  if (t >= total) return;
  int p = (int)(t % HW); int c = (int)((t / HW) % C);
  float v = x[t];
  if (noise) v += nw[0]*noise[p];
  v += cb[c];
  if (doAct) v = (v > 0.f ? v : 0.2f*v) * SQRT2F;
  x[t] = v;
}

// general upfirdn (true convolution: flipped kernel), supports channel slices & accumulation
__global__ void upfirdn_k(const float* __restrict__ in, long inB, long inOff,
                          float* __restrict__ out, long outB, long outOff,
                          int C, int H, int W, int OH, int OW,
                          int kid, int up, int down, int p0, float sign, int accum, long total) {
  long t = (long)blockIdx.x*256 + threadIdx.x;
  if (t >= total) return;
  int ox = (int)(t % OW); long r = t / OW;
  int oy = (int)(r % OH); r /= OH;
  int c  = (int)(r % C);  int b = (int)(r / C);
  const float* kp = (kid == 0) ? K_BLUR : &K_HAAR[kid-1][0];
  const int ks = (kid == 0) ? 4 : 2;
  const int dH = (H-1)*up + 1, dW = (W-1)*up + 1;
  const float* ip = in + (long)b*inB + inOff + (long)c*H*W;
  float acc = 0.f;
  for (int a = 0; a < ks; ++a) {
    int ty = oy*down + a - p0;
    if (ty < 0 || ty >= dH) continue;
    if (up == 2 && (ty & 1)) continue;
    int iy = (up == 2) ? (ty >> 1) : ty;
    for (int bb = 0; bb < ks; ++bb) {
      int tx = ox*down + bb - p0;
      if (tx < 0 || tx >= dW) continue;
      if (up == 2 && (tx & 1)) continue;
      int ix = (up == 2) ? (tx >> 1) : tx;
      acc += kp[(ks-1-a)*ks + (ks-1-bb)] * ip[(long)iy*W + ix];
    }
  }
  acc *= sign;
  float* op = out + (long)b*outB + outOff + ((long)c*OH + oy)*OW + ox;
  *op = accum ? (*op + acc) : acc;
}

// ---- host orchestration ---------------------------------------------------
extern "C" void kernel_launch(void* const* d_in, const int* in_sizes, int n_in,
                              void* d_out, int out_size, void* d_ws, size_t ws_size,
                              hipStream_t stream) {
  auto IN = [&](int i) -> const float* { return (const float*)d_in[i]; };
  bool jaxsorted = (n_in >= 1 && in_sizes[0] == 16);   // sorted pytree puts noises[0] (4x4) first

  struct SP { const float *w,*mw,*mb,*nw,*ab; };
  struct RP { const float *w,*mw,*mb,*bias; };
  const float* styles; const float* constp;
  const float *mlp_w[8], *mlp_b[8];
  SP conv1, convs[10]; RP rgb1, rgbs[5];
  const float* noi[11];

  if (!jaxsorted) {            // insertion order: styles, params{mlp,const,conv1,to_rgb1,convs,to_rgbs}, noises
    styles = IN(0);
    for (int l = 0; l < 8; ++l) { mlp_w[l] = IN(1+2*l); mlp_b[l] = IN(2+2*l); }
    constp = IN(17);
    conv1 = {IN(18),IN(19),IN(20),IN(21),IN(22)};
    rgb1  = {IN(23),IN(24),IN(25),IN(26)};
    for (int i = 0; i < 10; ++i) { int b = 27+5*i; convs[i] = {IN(b),IN(b+1),IN(b+2),IN(b+3),IN(b+4)}; }
    for (int i = 0; i < 5;  ++i) { int b = 77+4*i; rgbs[i]  = {IN(b),IN(b+1),IN(b+2),IN(b+3)}; }
    for (int l = 0; l < 11; ++l) noi[l] = IN(97+l);
  } else {                     // jax pytree (sorted keys): noises, params{const,conv1,convs,mlp,to_rgb1,to_rgbs}, styles
    for (int l = 0; l < 11; ++l) noi[l] = IN(l);
    constp = IN(11);
    conv1 = {IN(16),IN(14),IN(13),IN(15),IN(12)};
    for (int i = 0; i < 10; ++i) { int b = 17+5*i; convs[i] = {IN(b+4),IN(b+2),IN(b+1),IN(b+3),IN(b)}; }
    for (int l = 0; l < 8;  ++l) { mlp_b[l] = IN(67+2*l); mlp_w[l] = IN(68+2*l); }
    rgb1 = {IN(86),IN(85),IN(84),IN(83)};
    for (int i = 0; i < 5;  ++i) { int b = 87+4*i; rgbs[i] = {IN(b+3),IN(b+2),IN(b+1),IN(b)}; }
    styles = IN(107);
  }

  // workspace bump allocator
  char* base = (char*)d_ws; size_t off = 0;
  auto alloc = [&](size_t bytes) -> char* { char* p = base + off; off = (off + bytes + 255) & ~(size_t)255; return p; };
  float*    mapA = (float*)alloc(4096*4);
  float*    mapB = (float*)alloc(4096*4);
  _Float16* wlh  = (_Float16*)alloc((size_t)262144*2);
  float*    sbuf = (float*)alloc(4096*4);
  float*    dbuf = (float*)alloc(4096*4);
  float*    w2   = (float*)alloc((size_t)262144*4);
  _Float16* wh   = (_Float16*)alloc((size_t)9*262144*2);
  _Float16* xsh  = (_Float16*)alloc((size_t)8*512*127*127*2);
  float*    actA = (float*)alloc((size_t)8*512*129*129*4);
  float*    actB = (float*)alloc((size_t)8*512*129*129*4);
  float*    rgbA = (float*)alloc((size_t)8*12*128*128*4);
  float*    rgbB = (float*)alloc((size_t)8*12*128*128*4);
  float*    t3a  = (float*)alloc((size_t)8*3*256*256*4);
  float*    t3b  = (float*)alloc((size_t)8*3*256*256*4);
  (void)ws_size; (void)out_size;

  auto g1 = [&](long n) { return dim3((unsigned)((n + 255) / 256)); };
  const dim3 blk(256);

  // ---- mapping network ----
  pixel_norm_k<<<dim3(8), blk, 0, stream>>>(styles, mapA);
  float* mi = mapA; float* mo = mapB;
  for (int l = 0; l < 8; ++l) {
    prep_mapw_k<<<g1(262144), blk, 0, stream>>>(mlp_w[l], wlh);
    map_layer_k<<<dim3(4), blk, 0, stream>>>(mi, wlh, mlp_b[l], mo);
    float* t = mi; mi = mo; mo = t;
  }
  const float* wlat = mi;    // (8,512); identical for all 12 latents (repeat of w)

  auto conv_launch = [&](const _Float16* xin, const _Float16* wt, const float* dm, float* o,
                         int Cin, int Cout, int IH, int IW, int OH, int OW, int ks, int pad) {
    int tilesX = (OW + 15) / 16, tilesY = (OH + 7) / 8;
    conv_wmma_k<<<dim3((unsigned)(tilesX*tilesY), (unsigned)((Cout+63)/64), 8), blk, 0, stream>>>(
        xin, wt, dm, o, Cin, Cout, IH, IW, OH, OW, ks, pad, tilesX);
  };

  // ---- styled conv runner ----
  auto run_styled = [&](const float* xin, long xbs, int Cin, int Cout, int H, bool up,
                        const SP& p, const float* noise, float* outBuf, float* scratchBuf) {
    style_mod_k<<<g1((long)8*Cin), blk, 0, stream>>>(wlat, p.mw, p.mb, sbuf, Cin);
    prep_convw_k<<<g1((long)Cout*Cin), blk, 0, stream>>>(p.w, wh, w2, Cin, Cout, 3, up ? 1 : 0);
    demod_k<<<g1((long)8*Cout), blk, 0, stream>>>(w2, sbuf, dbuf, Cin, Cout);
    if (!up) {
      long tot = (long)8*Cin*H*H;
      mod_half_k<<<g1(tot), blk, 0, stream>>>(xin, xbs, sbuf, xsh, Cin, H*H, tot);
      conv_launch(xsh, wh, dbuf, outBuf, Cin, Cout, H, H, H, H, 3, 1);
      long t2 = (long)8*Cout*H*H;
      noise_act_k<<<g1(t2), blk, 0, stream>>>(outBuf, noise, p.nw, p.ab, Cout, H*H, 1, t2);
    } else {
      int D = 2*H - 1, O = 2*H + 1, F = 2*H;
      long tot = (long)8*Cin*D*D;
      dilate_mod_half_k<<<g1(tot), blk, 0, stream>>>(xin, sbuf, xsh, Cin, H, H, tot);
      conv_launch(xsh, wh, dbuf, scratchBuf, Cin, Cout, D, D, O, O, 3, 2);  // flipped W, pad=2: conv_transpose
      long t2 = (long)8*Cout*F*F;
      upfirdn_k<<<g1(t2), blk, 0, stream>>>(scratchBuf, (long)Cout*O*O, 0L,
          outBuf, (long)Cout*F*F, 0L, Cout, O, O, F, F, 0, 1, 1, 1, 1.0f, 0, t2); // blur pad(1,1)
      noise_act_k<<<g1(t2), blk, 0, stream>>>(outBuf, noise, p.nw, p.ab, Cout, F*F, 1, t2);
    }
  };

  // ---- to_rgb runner (1x1 modulated conv, no demod; Haar skip plumbing) ----
  auto run_rgb = [&](const float* xin, int Cin, int R, const RP& p,
                     float* outRgb, const float* skipPrev) {
    style_mod_k<<<g1((long)8*Cin), blk, 0, stream>>>(wlat, p.mw, p.mb, sbuf, Cin);
    prep_convw_k<<<g1((long)12*Cin), blk, 0, stream>>>(p.w, wh, w2, Cin, 12, 1, 0);
    long tot = (long)8*Cin*R*R;
    mod_half_k<<<g1(tot), blk, 0, stream>>>(xin, (long)Cin*R*R, sbuf, xsh, Cin, R*R, tot);
    conv_launch(xsh, wh, (const float*)nullptr, outRgb, Cin, 12, R, R, R, R, 1, 0);
    long t2 = (long)8*12*R*R;
    noise_act_k<<<g1(t2), blk, 0, stream>>>(outRgb, (const float*)nullptr, (const float*)nullptr,
                                            p.bias, 12, R*R, 0, t2);
    if (skipPrev) {
      int r = R/2, R2 = 2*R;
      const float sgn[4] = {1.f, -1.f, -1.f, 1.f};     // iwt: +ll, -lh, -hl, +hh
      long ti = (long)8*3*R*R;
      for (int q = 0; q < 4; ++q)                      // haar_iwt(skip): 12ch@r -> 3ch@R
        upfirdn_k<<<g1(ti), blk, 0, stream>>>(skipPrev, (long)12*r*r, (long)q*3*r*r,
            t3a, (long)3*R*R, 0L, 3, r, r, R, R, q+1, 2, 1, 1, sgn[q], q > 0 ? 1 : 0, ti);
      long tb = (long)8*3*R2*R2;                       // Upsample: blur up=2 pad(2,1): 3ch@R -> 3ch@2R
      upfirdn_k<<<g1(tb), blk, 0, stream>>>(t3a, (long)3*R*R, 0L,
          t3b, (long)3*R2*R2, 0L, 3, R, R, R2, R2, 0, 2, 1, 2, 1.0f, 0, tb);
      for (int q = 0; q < 4; ++q)                      // haar_dwt: 3ch@2R -> accumulate 12ch@R into out
        upfirdn_k<<<g1(ti), blk, 0, stream>>>(t3b, (long)3*R2*R2, 0L,
            outRgb, (long)12*R*R, (long)q*3*R*R, 3, R2, R2, R, R, q+1, 1, 2, 0, 1.0f, 1, ti);
    }
  };

  // ---- generator body ----
  float* cur = actA; float* alt = actB;
  run_styled(constp, 0L, 512, 512, 4, false, conv1, noi[0], cur, nullptr);  // const broadcast (xbs=0)
  run_rgb(cur, 512, 4, rgb1, rgbA, nullptr);
  float* skip = rgbA; float* rAlt = rgbB;

  static const int cins[10]  = {512,512,512,512,512,512,512,512,512,256};
  static const int couts[10] = {512,512,512,512,512,512,512,512,256,256};
  int H = 4;
  for (int j = 0; j < 5; ++j) {
    int ci0 = cins[2*j], co0 = couts[2*j];
    run_styled(cur, (long)ci0*H*H, ci0, co0, H, true, convs[2*j], noi[2*j+1], alt, cur);
    { float* t = cur; cur = alt; alt = t; }
    H *= 2;
    int ci1 = couts[2*j], co1 = couts[2*j+1];
    run_styled(cur, (long)ci1*H*H, ci1, co1, H, false, convs[2*j+1], noi[2*j+2], alt, nullptr);
    { float* t = cur; cur = alt; alt = t; }
    run_rgb(cur, co1, H, rgbs[j], rAlt, skip);
    { float* t = skip; skip = rAlt; rAlt = t; }
  }

  // final haar_iwt: skip 12ch@128 -> d_out 3ch@256
  {
    float* outp = (float*)d_out;
    const float sgn[4] = {1.f, -1.f, -1.f, 1.f};
    long ti = (long)8*3*256*256;
    for (int q = 0; q < 4; ++q)
      upfirdn_k<<<g1(ti), blk, 0, stream>>>(skip, (long)12*128*128, (long)q*3*128*128,
          outp, (long)3*256*256, 0L, 3, 128, 128, 256, 256, q+1, 2, 1, 1, sgn[q], q > 0 ? 1 : 0, ti);
  }
}
